// PoseNet_origional_mamba_74466142978505
// MI455X (gfx1250) — compile-verified, matrix operand order checked by reading, not run on hardware
//
#include <hip/hip_runtime.h>
#include <hip/hip_bf16.h>

// ---------------------------------------------------------------------------
// Types for CDNA5 WMMA / TDM (wave32)
// ---------------------------------------------------------------------------
typedef __attribute__((ext_vector_type(16))) __bf16 v16bf;
typedef __attribute__((ext_vector_type(4)))  __bf16 bf16x4;
typedef __attribute__((ext_vector_type(8)))  float  v8f;
typedef __attribute__((ext_vector_type(4)))  float  f32x4;
typedef __attribute__((ext_vector_type(4)))  unsigned int u32x4;
typedef __attribute__((ext_vector_type(8)))  int    i32x8;
typedef __attribute__((ext_vector_type(4)))  int    i32x4;

#define B_SZ     64
#define C_IN     3
#define HW       224
#define OHW      56
#define FE_OUT   512
#define D_MODEL  2048
#define D_STATE  16
#define D_CONV   4
#define D_INNER  4096
#define DT_RANK  128
#define KCONV    147          // 3*7*7
#define NPOS     (OHW*OHW)    // 3136
#define LTOK     128          // B*L tokens
#define LDS_STR  40           // LDS row pitch in bf16 (16B-aligned fragments)

// ---------------------------------------------------------------------------
// Helpers
// ---------------------------------------------------------------------------
__device__ __forceinline__ bf16x4 cvt4(f32x4 v) {
    bf16x4 r;                              // v_cvt_pk_bf16_f32
    r[0] = (__bf16)v[0]; r[1] = (__bf16)v[1];
    r[2] = (__bf16)v[2]; r[3] = (__bf16)v[3];
    return r;
}
__device__ __forceinline__ float softplusf(float v) {
    return v > 20.f ? v : log1pf(expf(v));
}
__device__ __forceinline__ float siluf(float v) {
    return v / (1.f + expf(-v));
}
__device__ __forceinline__ float epilogue(float v, const float* bias, int ng, int act) {
    if (bias) v += bias[ng];
    if (act == 1)      v = fmaxf(v, 0.f);
    else if (act == 2) v = softplusf(v);
    return v;
}

// Load a 16x32 bf16 fragment from an LDS tile (row-major, LDS_STR bf16/row).
__device__ __forceinline__ v16bf load_frag(const __bf16* lds, int row0, int lane) {
    int row = row0 + (lane & 15);
    int kk  = (lane & 16) ? 8 : 0;
    const u32x4* p0 = reinterpret_cast<const u32x4*>(lds + row * LDS_STR + kk);
    const u32x4* p1 = reinterpret_cast<const u32x4*>(lds + row * LDS_STR + kk + 16);
    union { u32x4 u[2]; v16bf v; } cvt;
    cvt.u[0] = *p0;
    cvt.u[1] = *p1;
    return cvt.v;
}

#define WMMA_BF16(a, b, c) \
    __builtin_amdgcn_wmma_f32_16x16x32_bf16(false, (a), false, (b), (short)0, (c), false, false)

// ---------------------------------------------------------------------------
// TDM: DMA one 64-row x 32-col bf16 tile (row pitch ldb elements) from global
// memory into LDS with a 40-bf16 destination pitch (pad 4 DWORDs every 16).
// D# layout per CDNA5 ISA.
// Toolchain uses the 6-arg builtin: (u32x4, i32x8, i32x4, i32x4, i32x8, i32).
// ---------------------------------------------------------------------------
__device__ __forceinline__ void tdm_load_tile(const __bf16* gsrc, int ldb,
                                              unsigned lds_addr) {
    unsigned long long ga = (unsigned long long)(uintptr_t)gsrc;
    u32x4 g0;
    g0[0] = 1u;                                     // count=1, user mode
    g0[1] = lds_addr;                               // LDS byte address
    g0[2] = (unsigned)(ga & 0xFFFFFFFFu);           // global_addr[31:0]
    g0[3] = (unsigned)((ga >> 32) & 0x01FFFFFFu)    // global_addr[56:32]
          | (2u << 30);                             // type = 2 ("image")
    i32x8 g1;
    g1[0] = (1 << 16)                               // data_size = 2 bytes
          | (1 << 20)                               // pad_enable
          | (3 << 22)                               // pad_interval: 16 DWORDs
          | (3 << 25);                              // pad_amount:   4 DWORDs
    g1[1] = (32 & 0xFFFF) << 16;                    // tensor_dim0 = 32 (lo16)
    g1[2] = (64 & 0xFFFF) << 16;                    // dim0 hi=0 | tensor_dim1 = 64
    g1[3] = (32 << 16);                             // dim1 hi=0 | tile_dim0 = 32
    g1[4] = 64;                                     // tile_dim1 = 64, tile_dim2 = 0
    g1[5] = ldb;                                    // tensor_dim0_stride[31:0]
    g1[6] = 0;
    g1[7] = 0;
    i32x4 z4 = {0, 0, 0, 0};
    i32x8 z8 = {0, 0, 0, 0, 0, 0, 0, 0};
    __builtin_amdgcn_tensor_load_to_lds(g0, g1, z4, z4, z8, 0);
}

// ---------------------------------------------------------------------------
// FAST GEMM:  C[M,N] = act( A[M,K](f32) * B[N,K](bf16)^T + bias[N] )
// Requires: M % 64 == 0 (via grid), N % 64 == 0, K % 32 == 0, lda % 4 == 0.
// A: f32 register staging + v_cvt_pk_bf16_f32 into LDS (ping-pong).
// B: bf16 tiles DMA'd by the Tensor Data Mover directly into LDS (ping-pong),
//    issued by wave 0, synchronized with s_wait_tensorcnt + barrier.
// 4 wave32s in 2x2; each wave owns a 32x32 subtile (4 WMMA accumulators).
// ---------------------------------------------------------------------------
__device__ __forceinline__ void g_loadA(const float* __restrict__ A, int lda,
                                        size_t m0, int k0, int tid, f32x4* rA) {
    #pragma unroll
    for (int c = 0; c < 4; ++c) {
        int idx = tid + 128 * c;            // 0..511 float4 chunks
        int r   = idx >> 3;                 // 64 rows
        int kq  = (idx & 7) << 2;           // 0,4,...,28
        rA[c] = *(const f32x4*)(A + (m0 + r) * lda + k0 + kq);
    }
}
__device__ __forceinline__ void s_storeA(__bf16* lA, int tid, const f32x4* rA) {
    #pragma unroll
    for (int c = 0; c < 4; ++c) {
        int idx = tid + 128 * c;
        int r   = idx >> 3;
        int kq  = (idx & 7) << 2;
        *(bf16x4*)(lA + r * LDS_STR + kq) = cvt4(rA[c]);   // ds_store_b64
    }
}

__global__ __launch_bounds__(128)
void gemm_a32_btdm(const float* __restrict__ A, int lda,
                   const __bf16* __restrict__ B, int ldb,
                   float* __restrict__ C, int ldc, int K,
                   const float* __restrict__ bias, int act) {
    __shared__ __align__(16) __bf16 lA[2][64 * LDS_STR];
    __shared__ __align__(16) __bf16 lB[2][64 * LDS_STR];

    const int tid  = threadIdx.x;
    const int lane = tid & 31;
    const int w    = tid >> 5;
    const int wm   = w >> 1;
    const int wn   = w & 1;
    const size_t m0 = (size_t)blockIdx.y * 64;
    const size_t n0 = (size_t)blockIdx.x * 64;

    const unsigned ldsB0 = (unsigned)(uintptr_t)&lB[0][0];
    const unsigned ldsB1 = (unsigned)(uintptr_t)&lB[1][0];
    const __bf16* gB = B + n0 * ldb;

    v8f acc00 = {0.f,0.f,0.f,0.f,0.f,0.f,0.f,0.f};
    v8f acc01 = acc00, acc10 = acc00, acc11 = acc00;

    f32x4 rA[4];
    g_loadA(A, lda, m0, 0, tid, rA);
    s_storeA(lA[0], tid, rA);
    if (w == 0) tdm_load_tile(gB, ldb, ldsB0);        // TENSORcnt++

    int cur = 0;
    for (int k0 = 0; k0 < K; k0 += 32) {
        if (w == 0) __builtin_amdgcn_s_wait_tensorcnt(0);
        __syncthreads();                              // tile `cur` visible
        const bool more = (k0 + 32) < K;
        if (more) {
            __builtin_prefetch(A + (m0 + lane) * lda + k0 + 64, 0, 1);
            g_loadA(A, lda, m0, k0 + 32, tid, rA);
            if (w == 0)
                tdm_load_tile(gB + (k0 + 32), ldb, cur ? ldsB0 : ldsB1);
        }
        v16bf a0 = load_frag(lA[cur], wm * 32,      lane);
        v16bf a1 = load_frag(lA[cur], wm * 32 + 16, lane);
        v16bf b0 = load_frag(lB[cur], wn * 32,      lane);
        v16bf b1 = load_frag(lB[cur], wn * 32 + 16, lane);
        acc00 = WMMA_BF16(a0, b0, acc00);
        acc01 = WMMA_BF16(a0, b1, acc01);
        acc10 = WMMA_BF16(a1, b0, acc10);
        acc11 = WMMA_BF16(a1, b1, acc11);
        if (more) s_storeA(lA[cur ^ 1], tid, rA);
        cur ^= 1;
    }

    const int rowHalf = (lane & 16) ? 8 : 0;
    const int colLane = lane & 15;
    #pragma unroll
    for (int i = 0; i < 2; ++i) {
        #pragma unroll
        for (int j = 0; j < 2; ++j) {
            const v8f& a = (i == 0) ? (j == 0 ? acc00 : acc01)
                                    : (j == 0 ? acc10 : acc11);
            int ng = (int)n0 + wn * 32 + j * 16 + colLane;
            #pragma unroll
            for (int r = 0; r < 8; ++r) {
                size_t mg = m0 + wm * 32 + i * 16 + r + rowHalf;
                C[mg * ldc + ng] = epilogue(a[r], bias, ng, act);
            }
        }
    }
}

// ---------------------------------------------------------------------------
// Generic guarded GEMM (only used for x_dbl, N=160). 32x32 tile, 4 waves.
// ---------------------------------------------------------------------------
__global__ __launch_bounds__(128)
void gemm_bf16_wmma(const float* __restrict__ A, int lda,
                    const float* __restrict__ B, int ldb,
                    float* __restrict__ C, int ldc,
                    int M, int N, int K,
                    const float* __restrict__ bias, int act) {
    __shared__ __align__(16) __bf16 lA[32 * LDS_STR];
    __shared__ __align__(16) __bf16 lB[32 * LDS_STR];

    const int tid  = threadIdx.x;
    const int lane = tid & 31;
    const int w    = tid >> 5;
    const int wm   = w >> 1;
    const int wn   = w & 1;
    const int m0   = blockIdx.y * 32;
    const int n0   = blockIdx.x * 32;

    v8f acc = {0.f,0.f,0.f,0.f,0.f,0.f,0.f,0.f};

    for (int k0 = 0; k0 < K; k0 += 32) {
        #pragma unroll
        for (int j = 0; j < 8; ++j) {
            int e  = tid + 128 * j;
            int r  = e >> 5;
            int kk = e & 31;
            int k  = k0 + kk;
            int mg = m0 + r;
            float va = (mg < M && k < K) ? A[(size_t)mg * lda + k] : 0.f;
            lA[r * LDS_STR + kk] = (__bf16)va;
            int ng = n0 + r;
            float vb = (ng < N && k < K) ? B[(size_t)ng * ldb + k] : 0.f;
            lB[r * LDS_STR + kk] = (__bf16)vb;
        }
        __syncthreads();
        v16bf af = load_frag(lA, wm * 16, lane);
        v16bf bf = load_frag(lB, wn * 16, lane);
        acc = WMMA_BF16(af, bf, acc);
        __syncthreads();
    }

    const int ng = n0 + wn * 16 + (lane & 15);
    #pragma unroll
    for (int r = 0; r < 8; ++r) {
        int mg = m0 + wm * 16 + r + ((lane & 16) ? 8 : 0);
        if (mg < M && ng < N)
            C[(size_t)mg * ldc + ng] = epilogue(acc[r], bias, ng, act);
    }
}

// ---------------------------------------------------------------------------
// Conv stem as implicit GEMM, fused bias+ReLU+spatial mean.
//   M = 3136 positions (per image), N = 512 channels, K = 147 (padded 160).
// ---------------------------------------------------------------------------
__device__ __forceinline__ void conv_stage(const float* __restrict__ xb,
                                           const float* __restrict__ wrow,
                                           __bf16* lA, __bf16* lB,
                                           int arow, int khalf, int k0,
                                           int ih0, int iw0, bool kguard) {
    int ks = k0 + khalf;
    int c  = ks / 49;
    int rm = ks - c * 49;
    int kh = rm / 7;
    int kw = rm - kh * 7;
    union { __bf16 h[16]; u32x4 q[2]; } ta, tb;
    #pragma unroll
    for (int j = 0; j < 16; ++j) {
        int k  = ks + j;
        int ih = ih0 + kh, iw = iw0 + kw;
        float va = 0.f;
        if ((!kguard || k < KCONV) && ((ih | iw) >= 0))
            va = xb[((size_t)c * HW + ih) * HW + iw];
        ta.h[j] = (__bf16)va;
        float vb = (!kguard || k < KCONV) ? wrow[k] : 0.f;
        tb.h[j] = (__bf16)vb;
        if (++kw == 7) { kw = 0; if (++kh == 7) { kh = 0; ++c; } }
    }
    u32x4* pA = (u32x4*)(lA + arow * LDS_STR + khalf);   // 16B aligned
    pA[0] = ta.q[0]; pA[1] = ta.q[1];                    // 2x ds_store_b128
    u32x4* pB = (u32x4*)(lB + arow * LDS_STR + khalf);
    pB[0] = tb.q[0]; pB[1] = tb.q[1];
}

__global__ __launch_bounds__(128)
void conv_stem_wmma(const float* __restrict__ x,
                    const float* __restrict__ cw,
                    const float* __restrict__ cb,
                    float* __restrict__ f) {
    __shared__ __align__(16) __bf16 lA[64 * LDS_STR];
    __shared__ __align__(16) __bf16 lB[64 * LDS_STR];
    __shared__ float csum[64];

    const int tid  = threadIdx.x;
    const int lane = tid & 31;
    const int w    = tid >> 5;
    const int wm   = w >> 1;
    const int wn   = w & 1;
    const int b    = blockIdx.z;
    const int m0   = blockIdx.y * 64;   // 3136/64 = 49 exact
    const int n0   = blockIdx.x * 64;   // 512/64  = 8 exact

    const int arow  = tid >> 1;
    const int khalf = (tid & 1) * 16;
    const int m     = m0 + arow;
    const int ih0   = (m / OHW) * 4 - 3;
    const int iw0   = (m % OHW) * 4 - 3;
    const float* xb   = x  + (size_t)b * C_IN * HW * HW;
    const float* wrow = cw + (size_t)(n0 + arow) * KCONV;

    v8f acc00 = {0.f,0.f,0.f,0.f,0.f,0.f,0.f,0.f};
    v8f acc01 = acc00, acc10 = acc00, acc11 = acc00;

    #pragma unroll 1
    for (int k0 = 0; k0 < 160; k0 += 32) {
        conv_stage(xb, wrow, lA, lB, arow, khalf, k0, ih0, iw0,
                   /*kguard=*/k0 == 128);
        __syncthreads();
        v16bf a0 = load_frag(lA, wm * 32,      lane);
        v16bf a1 = load_frag(lA, wm * 32 + 16, lane);
        v16bf b0 = load_frag(lB, wn * 32,      lane);
        v16bf b1 = load_frag(lB, wn * 32 + 16, lane);
        acc00 = WMMA_BF16(a0, b0, acc00);
        acc01 = WMMA_BF16(a0, b1, acc01);
        acc10 = WMMA_BF16(a1, b0, acc10);
        acc11 = WMMA_BF16(a1, b1, acc11);
        __syncthreads();
    }

    if (tid < 64) csum[tid] = 0.f;
    __syncthreads();
    const int colLane = lane & 15;
    #pragma unroll
    for (int j = 0; j < 2; ++j) {
        int cg  = wn * 32 + j * 16 + colLane;
        float bsv = cb[n0 + cg];
        float s = 0.f;
        #pragma unroll
        for (int r = 0; r < 8; ++r) {
            s += fmaxf(((j == 0) ? acc00[r] : acc01[r]) + bsv, 0.f);
            s += fmaxf(((j == 0) ? acc10[r] : acc11[r]) + bsv, 0.f);
        }
        atomicAdd(&csum[cg], s);                // ds_add_f32
    }
    __syncthreads();
    if (tid < 64)
        atomicAdd(&f[(size_t)b * FE_OUT + n0 + tid],
                  csum[tid] * (1.f / (float)NPOS));
}

// ---------------------------------------------------------------------------
// Small elementwise kernels
// ---------------------------------------------------------------------------
__global__ void zero_kernel(float* p, int n) {
    for (int i = blockIdx.x * blockDim.x + threadIdx.x; i < n;
         i += gridDim.x * blockDim.x) p[i] = 0.f;
}

// f32 -> bf16 weight pre-conversion (vectorized, n % 4 == 0)
__global__ void cvt_f32_bf16(const float* __restrict__ in,
                             __bf16* __restrict__ out, int n4) {
    for (int i = blockIdx.x * blockDim.x + threadIdx.x; i < n4;
         i += gridDim.x * blockDim.x)
        ((bf16x4*)out)[i] = cvt4(((const f32x4*)in)[i]);
}

// seq[b][0][d] = ffc[b][2047-d] ; seq[b][1][d] = ffc[b][d]
__global__ void build_seq(const float* __restrict__ ffc, float* __restrict__ seq) {
    int i = blockIdx.x * blockDim.x + threadIdx.x;
    if (i >= LTOK * D_MODEL) return;
    int d   = i & (D_MODEL - 1);
    int row = i / D_MODEL;
    int b   = row >> 1;
    int l   = row & 1;
    int src = (l == 0) ? (D_MODEL - 1 - d) : d;
    seq[i] = ffc[(size_t)b * D_MODEL + src];
}

// depthwise causal conv1d (L=2, taps 4) + bias + SiLU on the x-half of xz
__global__ void conv1d_silu(const float* __restrict__ xz,
                            const float* __restrict__ cw,
                            const float* __restrict__ cb,
                            float* __restrict__ xsi) {
    int i = blockIdx.x * blockDim.x + threadIdx.x;
    if (i >= B_SZ * D_INNER) return;
    int e = i & (D_INNER - 1);
    int b = i / D_INNER;
    size_t r0 = (size_t)(b * 2) * (2 * D_INNER);
    size_t r1 = (size_t)(b * 2 + 1) * (2 * D_INNER);
    float x0 = xz[r0 + e];
    float x1 = xz[r1 + e];
    float w2 = cw[e * D_CONV + 2];
    float w3 = cw[e * D_CONV + 3];
    float bb = cb[e];
    xsi[(size_t)(b * 2)     * D_INNER + e] = siluf(w3 * x0 + bb);
    xsi[(size_t)(b * 2 + 1) * D_INNER + e] = siluf(w2 * x0 + w3 * x1 + bb);
}

// L=2 selective scan, fused with D-skip and SiLU(z) gate.
__global__ void ssm_scan(const float* __restrict__ xsi,
                         const float* __restrict__ xdbl,   // [128,160]
                         const float* __restrict__ dt,     // softplus'd [128,4096]
                         const float* __restrict__ A_log,  // [4096,16]
                         const float* __restrict__ Dvec,
                         const float* __restrict__ xz,
                         float* __restrict__ ybl) {
    int i = blockIdx.x * blockDim.x + threadIdx.x;
    if (i >= B_SZ * D_INNER) return;
    int e = i & (D_INNER - 1);
    int b = i / D_INNER;
    int t0 = b * 2, t1 = b * 2 + 1;

    float x0  = xsi[(size_t)t0 * D_INNER + e];
    float x1  = xsi[(size_t)t1 * D_INNER + e];
    float dt0 = dt[(size_t)t0 * D_INNER + e];
    float dt1 = dt[(size_t)t1 * D_INNER + e];
    const float* xd0 = xdbl + (size_t)t0 * 160;
    const float* xd1 = xdbl + (size_t)t1 * 160;

    float y0 = 0.f, y1 = 0.f;
    #pragma unroll
    for (int n = 0; n < D_STATE; ++n) {
        float Aen = -expf(A_log[(size_t)e * D_STATE + n]);
        float h   = dt0 * xd0[DT_RANK + n] * x0;
        y0 += h * xd0[DT_RANK + D_STATE + n];
        h = h * expf(dt1 * Aen) + dt1 * xd1[DT_RANK + n] * x1;
        y1 += h * xd1[DT_RANK + D_STATE + n];
    }
    float dsk = Dvec[e];
    y0 += x0 * dsk;
    y1 += x1 * dsk;
    float z0 = xz[(size_t)t0 * (2 * D_INNER) + D_INNER + e];
    float z1 = xz[(size_t)t1 * (2 * D_INNER) + D_INNER + e];
    ybl[(size_t)t0 * D_INNER + e] = y0 * siluf(z0);
    ybl[(size_t)t1 * D_INNER + e] = y1 * siluf(z1);
}

// final heads
__global__ void heads_kernel(const float* __restrict__ mout,
                             const float* __restrict__ wx, const float* __restrict__ bx,
                             const float* __restrict__ ww, const float* __restrict__ bw,
                             float* __restrict__ out) {
    __shared__ float red[256];
    int b   = blockIdx.x;
    int tid = threadIdx.x;
    const float* feat = mout + (size_t)(b * 2 + 1) * D_MODEL;
    for (int j = 0; j < 7; ++j) {
        const float* wr = (j < 3) ? wx + (size_t)j * D_MODEL
                                  : ww + (size_t)(j - 3) * D_MODEL;
        float bias = (j < 3) ? bx[j] : bw[j - 3];
        float s = 0.f;
        for (int d = tid; d < D_MODEL; d += 256) s += feat[d] * wr[d];
        red[tid] = s;
        __syncthreads();
        for (int st = 128; st > 0; st >>= 1) {
            if (tid < st) red[tid] += red[tid + st];
            __syncthreads();
        }
        if (tid == 0) out[b * 7 + j] = red[0] + bias;
        __syncthreads();
    }
}

// ---------------------------------------------------------------------------
// Host side
// ---------------------------------------------------------------------------
extern "C" void kernel_launch(void* const* d_in, const int* in_sizes, int n_in,
                              void* d_out, int out_size, void* d_ws, size_t ws_size,
                              hipStream_t stream) {
    const float* x        = (const float*)d_in[0];
    const float* conv_w   = (const float*)d_in[1];
    const float* conv_b   = (const float*)d_in[2];
    const float* fc_w     = (const float*)d_in[3];
    const float* fc_b     = (const float*)d_in[4];
    const float* W_in     = (const float*)d_in[5];
    const float* conv1d_w = (const float*)d_in[6];
    const float* conv1d_b = (const float*)d_in[7];
    const float* W_xproj  = (const float*)d_in[8];
    const float* W_dt     = (const float*)d_in[9];
    const float* b_dt     = (const float*)d_in[10];
    const float* A_log    = (const float*)d_in[11];
    const float* Dvec     = (const float*)d_in[12];
    const float* W_out    = (const float*)d_in[13];
    const float* fc_xyz_w = (const float*)d_in[14];
    const float* fc_xyz_b = (const float*)d_in[15];
    const float* fc_wpqr_w= (const float*)d_in[16];
    const float* fc_wpqr_b= (const float*)d_in[17];
    float* out = (float*)d_out;

    // workspace layout (floats)
    float* ws   = (float*)d_ws;
    float* f    = ws;                        // 64*512
    float* ffc  = f    + B_SZ * FE_OUT;      // 64*2048
    float* seq  = ffc  + B_SZ * D_MODEL;     // 128*2048
    float* xz   = seq  + LTOK * D_MODEL;     // 128*8192
    float* xsi  = xz   + LTOK * 2 * D_INNER; // 128*4096
    float* xdbl = xsi  + LTOK * D_INNER;     // 128*160
    float* dtr  = xdbl + LTOK * 160;         // 128*4096
    float* ybl  = dtr  + LTOK * D_INNER;     // 128*4096
    float* mout = ybl  + LTOK * D_INNER;     // 128*2048
    // bf16 weight copies (element counts halved when viewed as floats)
    float* p    = mout + LTOK * D_MODEL;
    __bf16* bW_in  = (__bf16*)p;  p += (2 * D_INNER) * D_MODEL / 2;
    __bf16* bW_out = (__bf16*)p;  p += D_MODEL * D_INNER / 2;
    __bf16* bfc_w  = (__bf16*)p;  p += D_MODEL * FE_OUT / 2;
    __bf16* bW_dt  = (__bf16*)p;  p += D_INNER * DT_RANK / 2;

    // 0) one-time (per launch) weight conversions f32 -> bf16
    cvt_f32_bf16<<<2048, 256, 0, stream>>>(W_in,  bW_in,  (2*D_INNER)*D_MODEL/4);
    cvt_f32_bf16<<<2048, 256, 0, stream>>>(W_out, bW_out, D_MODEL*D_INNER/4);
    cvt_f32_bf16<<<512,  256, 0, stream>>>(fc_w,  bfc_w,  D_MODEL*FE_OUT/4);
    cvt_f32_bf16<<<512,  256, 0, stream>>>(W_dt,  bW_dt,  D_INNER*DT_RANK/4);

    // 1) conv stem (implicit GEMM + fused mean)
    zero_kernel<<<64, 256, 0, stream>>>(f, B_SZ * FE_OUT);
    conv_stem_wmma<<<dim3(FE_OUT / 64, NPOS / 64, B_SZ), 128, 0, stream>>>(
        x, conv_w, conv_b, f);

    // 2) fc: ffc = relu(f @ fc_w^T + fc_b)   [64,512]x[2048,512]
    gemm_a32_btdm<<<dim3(D_MODEL / 64, 1), 128, 0, stream>>>(
        f, FE_OUT, bfc_w, FE_OUT, ffc, D_MODEL, FE_OUT, fc_b, 1);

    // 3) seq = stack([reverse(ffc), ffc])
    build_seq<<<(LTOK * D_MODEL + 255) / 256, 256, 0, stream>>>(ffc, seq);

    // 4) xz = seq @ W_in^T   [128,2048]x[8192,2048]
    gemm_a32_btdm<<<dim3(2 * D_INNER / 64, LTOK / 64), 128, 0, stream>>>(
        seq, D_MODEL, bW_in, D_MODEL, xz, 2 * D_INNER, D_MODEL, nullptr, 0);

    // 5) depthwise conv1d + SiLU
    conv1d_silu<<<(B_SZ * D_INNER + 255) / 256, 256, 0, stream>>>(
        xz, conv1d_w, conv1d_b, xsi);

    // 6) x_dbl = xsi @ W_xproj^T   [128,4096]x[160,4096]  (N=160 -> guarded path)
    gemm_bf16_wmma<<<dim3(160 / 32, LTOK / 32), 128, 0, stream>>>(
        xsi, D_INNER, W_xproj, D_INNER, xdbl, 160,
        LTOK, 160, D_INNER, nullptr, 0);

    // 7) dt = softplus(x_dbl[:, :128] @ W_dt^T + b_dt)  [128,128]x[4096,128]
    gemm_a32_btdm<<<dim3(D_INNER / 64, LTOK / 64), 128, 0, stream>>>(
        xdbl, 160, bW_dt, DT_RANK, dtr, D_INNER, DT_RANK, b_dt, 2);

    // 8) L=2 selective scan + D skip + SiLU(z) gate
    ssm_scan<<<(B_SZ * D_INNER + 255) / 256, 256, 0, stream>>>(
        xsi, xdbl, dtr, A_log, Dvec, xz, ybl);

    // 9) mout = ybl @ W_out^T   [128,4096]x[2048,4096]
    gemm_a32_btdm<<<dim3(D_MODEL / 64, LTOK / 64), 128, 0, stream>>>(
        ybl, D_INNER, bW_out, D_INNER, mout, D_MODEL, D_INNER, nullptr, 0);

    // 10) heads
    heads_kernel<<<B_SZ, 256, 0, stream>>>(
        mout, fc_xyz_w, fc_xyz_b, fc_wpqr_w, fc_wpqr_b, out);
}
